// Hosvd_35150012351014
// MI455X (gfx1250) — matrix operand order repaired and checked by reading, not compiled, not run on hardware
//
#include <hip/hip_runtime.h>
#include <hip/hip_bf16.h>

// MI455X / gfx1250 fused HOSVD-cell kernel, round 3.
// Two-stage multilinear contraction, stage 1 on v_wmma_f32_16x16x32_bf16:
//   Y[b,q01,r] = sum_q23 h23[b,q23] * T[a,q01,q23,r]   (GEMM, N=672, K=96)
//   ris[b,r]   = sum_q01 h01[b,q01] * Y[b,q01,r]       (in-register reduce)
// A-fragments (h23) are built ONCE per wave; reduction reads a transposed
// h01t[a][q01][b] LDS table with 2 ds_load_b128 + 8 FMA per n-tile and one
// ds_swizzle (lane^8) even/odd-q01 combine at the end. No guards in hot loop.

typedef __bf16 v16bf __attribute__((ext_vector_type(16)));
typedef float  v8f   __attribute__((ext_vector_type(8)));

#define NTILE  42                       // 672 N-cols = 84 q01 (pad 81) x 8 r
#define OFF_T  0                        // 4*42*3 = 504 frags * 1KB
#define OFF_A  (504*1024)               // 64 frags (U)
#define OFF_O  (OFF_A + 64*1024)        // 64 frags (U_out)
// total ws use: 647,168 bytes

union AFrag { v16bf v; __bf16 e[16]; };
union BFrag { v16bf v; uint4 q[2]; };

// ---------------------------------------------------------------------------
// Repack T, U, U_out into 16x16x32 bf16 B-operand lane layout:
// lane (hi=lane>>4, n=lane&15) holds column n, K = 16*hi + e, e in [0,16).
// ---------------------------------------------------------------------------
__global__ void hosvd_repack(const float* __restrict__ U,
                             const float* __restrict__ Uo,
                             const float* __restrict__ T,
                             __hip_bfloat16* __restrict__ ws) {
  int gid  = blockIdx.x * blockDim.x + threadIdx.x;
  int frag = gid >> 5, lane = gid & 31;
  if (frag >= 632) return;
  int hi = lane >> 4, n = lane & 15;
  __hip_bfloat16* dst = ws + (size_t)frag * 512 + lane * 16;

  if (frag < 504) {            // T[a]: B[K=q23, N=(q01,r)] chunks
    int a = frag / 126, rem = frag % 126, nt = rem / 3, s = rem % 3;
    int ncol = nt * 16 + n, q01 = ncol >> 3, r = ncol & 7;
    #pragma unroll
    for (int i = 0; i < 16; ++i) {
      int q23 = s * 32 + hi * 16 + i;
      float v = (q01 < 81 && q23 < 81)
                  ? T[a * 52488 + (q01 * 81 + q23) * 8 + r] : 0.f;
      dst[i] = __float2bfloat16(v);
    }
  } else if (frag < 568) {     // U[c] as [256, 32] -> 2 ntiles x 8 kchunks
    int f = frag - 504; int c = f >> 4, nt = (f >> 3) & 1, kk = f & 7;
    int ncol = nt * 16 + n, ai = ncol >> 3, r = ncol & 7;
    #pragma unroll
    for (int i = 0; i < 16; ++i) {
      int krow = kk * 32 + hi * 16 + i;
      dst[i] = __float2bfloat16(U[((c * 4 + ai) * 256 + krow) * 8 + r]);
    }
  } else {                     // U_out[a] as [8->32, 256] -> 16 ntiles
    int f = frag - 568; int a = f >> 4, nt = f & 15;
    int ncol = nt * 16 + n;
    #pragma unroll
    for (int i = 0; i < 16; ++i) {
      int K = hi * 16 + i;
      float v = (K < 8) ? Uo[(a * 8 + K) * 256 + ncol] : 0.f;
      dst[i] = __float2bfloat16(v);
    }
  }
}

// ---------------------------------------------------------------------------
// Fused main kernel: 512 blocks x 128 threads; block owns 16 batch rows.
// Wave = child c in phase 0, aggregator a in phases 1-2.
// ---------------------------------------------------------------------------
__global__ __launch_bounds__(128) void hosvd_main(
    const float* __restrict__ nh,     // [8192,4,256]
    const float* __restrict__ bbias,  // [4,4,1,8] -> [c][a*8+r]
    const float* __restrict__ bout,   // [4,1,256]
    const __hip_bfloat16* __restrict__ ws,
    float* __restrict__ out) {        // [8192, 4*256]
  __shared__ float xs   [16 * 128];       // x[b16][c][a*8+r]           8 KB
  __shared__ float sh01t[4 * 84 * 16];    // h01t[a][q01(pad84)][b]    21 KB
  __shared__ float sh23 [16 * 4 * 96];    // h23[b][a][q23(pad96)]     24 KB
  __shared__ float ptmp [4 * 16 * 16];    // ris scratch                4 KB

  const int tid  = threadIdx.x;
  const int wave = tid >> 5, lane = tid & 31;
  const int hi   = lane >> 4, M = lane & 15, n = lane & 15;
  const int b0   = blockIdx.x * 16;

  // ---- Phase 0: x[:, c=wave, :, :] = nh @ U + b  (bf16 WMMA, K=256) ----
  {
    const int c = wave;
    v8f acc0 = {}, acc1 = {};
    const float* row = nh + (size_t)(b0 + M) * 1024 + c * 256 + hi * 16;
    const char* ba = (const char*)ws + OFF_A + (size_t)(c * 2) * 8 * 1024;
    #pragma unroll
    for (int kk = 0; kk < 8; ++kk) {
      AFrag af;
      const float4* p = (const float4*)(row + kk * 32);
      float4 f0 = p[0], f1 = p[1], f2 = p[2], f3 = p[3];
      af.e[0]=(__bf16)f0.x; af.e[1]=(__bf16)f0.y; af.e[2]=(__bf16)f0.z; af.e[3]=(__bf16)f0.w;
      af.e[4]=(__bf16)f1.x; af.e[5]=(__bf16)f1.y; af.e[6]=(__bf16)f1.z; af.e[7]=(__bf16)f1.w;
      af.e[8]=(__bf16)f2.x; af.e[9]=(__bf16)f2.y; af.e[10]=(__bf16)f2.z; af.e[11]=(__bf16)f2.w;
      af.e[12]=(__bf16)f3.x; af.e[13]=(__bf16)f3.y; af.e[14]=(__bf16)f3.z; af.e[15]=(__bf16)f3.w;
      BFrag bf0, bf1;
      const uint4* p0 = (const uint4*)(ba + (size_t)kk * 1024 + lane * 32);
      const uint4* p1 = (const uint4*)(ba + (size_t)(8 + kk) * 1024 + lane * 32);
      bf0.q[0] = p0[0]; bf0.q[1] = p0[1];
      bf1.q[0] = p1[0]; bf1.q[1] = p1[1];
      acc0 = __builtin_amdgcn_wmma_f32_16x16x32_bf16(false, af.v, false, bf0.v,
                                                     (short)0, acc0, false, false);
      acc1 = __builtin_amdgcn_wmma_f32_16x16x32_bf16(false, af.v, false, bf1.v,
                                                     (short)0, acc1, false, false);
    }
    const float biasA = bbias[c * 32 + n];
    const float biasB = bbias[c * 32 + 16 + n];
    #pragma unroll
    for (int j = 0; j < 8; ++j) {            // C/D: elem j -> M=j+8*hi, N=lane&15
      int m = j + 8 * hi;
      xs[m * 128 + c * 32 + n]      = acc0[j] + biasA;
      xs[m * 128 + c * 32 + 16 + n] = acc1[j] + biasB;
    }
  }
  __syncthreads();

  // ---- Phase 0.5: tables h01t[a][q01][b] (pad 84) and h23[b][a][q23] (96) --
  for (int idx = tid; idx < 4 * 84 * 16; idx += 128) {
    int b   = idx & 15;
    int q01 = (idx >> 4) % 84;
    int aa  = idx / (84 * 16);
    float v = 0.f;
    if (q01 < 81) {
      int qa = q01 / 9, qb = q01 % 9;
      float v0 = (qa < 8) ? xs[b * 128 + 0 * 32 + aa * 8 + qa] : 1.f;
      float v1 = (qb < 8) ? xs[b * 128 + 1 * 32 + aa * 8 + qb] : 1.f;
      v = v0 * v1;
    }
    sh01t[idx] = v;
  }
  for (int idx = tid; idx < 16 * 4 * 96; idx += 128) {
    int q  = idx % 96;  int t = idx / 96;
    int aa = t & 3,  bb = t >> 2;
    float v = 0.f;
    if (q < 81) {
      int qa = q / 9, qb = q % 9;
      float v2 = (qa < 8) ? xs[bb * 128 + 2 * 32 + aa * 8 + qa] : 1.f;
      float v3 = (qb < 8) ? xs[bb * 128 + 3 * 32 + aa * 8 + qb] : 1.f;
      v = v2 * v3;
    }
    sh23[idx] = v;
  }
  __syncthreads();

  // ---- Phase 1: Y = h23 @ T[a] (bf16 WMMA, K=96, N=672) + q01 reduce ----
  {
    const int a = wave;
    // A-fragments (h23) built once: 3 chunks of K=32
    const float* h23row = &sh23[(M * 4 + a) * 96 + hi * 16];
    AFrag af[3];
    #pragma unroll
    for (int s = 0; s < 3; ++s) {
      const float4* hp = (const float4*)(h23row + s * 32);
      float4 g0 = hp[0], g1 = hp[1], g2 = hp[2], g3 = hp[3];
      af[s].e[0]=(__bf16)g0.x; af[s].e[1]=(__bf16)g0.y;
      af[s].e[2]=(__bf16)g0.z; af[s].e[3]=(__bf16)g0.w;
      af[s].e[4]=(__bf16)g1.x; af[s].e[5]=(__bf16)g1.y;
      af[s].e[6]=(__bf16)g1.z; af[s].e[7]=(__bf16)g1.w;
      af[s].e[8]=(__bf16)g2.x; af[s].e[9]=(__bf16)g2.y;
      af[s].e[10]=(__bf16)g2.z; af[s].e[11]=(__bf16)g2.w;
      af[s].e[12]=(__bf16)g3.x; af[s].e[13]=(__bf16)g3.y;
      af[s].e[14]=(__bf16)g3.z; af[s].e[15]=(__bf16)g3.w;
    }
    float partial[8] = {0.f,0.f,0.f,0.f,0.f,0.f,0.f,0.f};
    const char* tb = (const char*)ws + OFF_T + (size_t)a * 126 * 1024 +
                     lane * 32;
    const int hi8 = n >> 3;                   // this lane's q01 parity
    for (int g = 0; g < 6; ++g) {
      #pragma unroll
      for (int u = 0; u < 7; ++u) {
        const int nt = g * 7 + u;
        v8f acc = {};
        #pragma unroll
        for (int s = 0; s < 3; ++s) {
          BFrag bf;
          const uint4* bp = (const uint4*)(tb + (size_t)(nt * 3 + s) * 1024);
          bf.q[0] = bp[0]; bf.q[1] = bp[1];
          acc = __builtin_amdgcn_wmma_f32_16x16x32_bf16(
                    false, af[s].v, false, bf.v, (short)0, acc, false, false);
        }
        // acc elem j = Y[b=j+8*hi, col=nt*16+n]; q01 = 2*nt + hi8, r = n&7
        const int q01 = nt * 2 + hi8;
        const float4* hq =
            (const float4*)(&sh01t[(a * 84 + q01) * 16 + hi * 8]);
        float4 h0v = hq[0], h1v = hq[1];
        partial[0] += h0v.x * acc[0]; partial[1] += h0v.y * acc[1];
        partial[2] += h0v.z * acc[2]; partial[3] += h0v.w * acc[3];
        partial[4] += h1v.x * acc[4]; partial[5] += h1v.y * acc[5];
        partial[6] += h1v.z * acc[6]; partial[7] += h1v.w * acc[7];
      }
    }
    // combine even/odd q01 halves: lane n with lane n^8 (ds_swizzle SWAPX8)
    #pragma unroll
    for (int j = 0; j < 8; ++j) {
      int sw = __builtin_amdgcn_ds_swizzle(
                   __builtin_bit_cast(int, partial[j]), 0x201F);
      float sum = partial[j] + __builtin_bit_cast(float, sw);
      if (n < 8)                               // ris[b=j+8*hi][r=n]
        ptmp[(a * 16 + j + 8 * hi) * 16 + n] = sum;
    }
  }
  __syncthreads();

  // ---- Phase 2: out = ris @ U_out + b_out  (K=8 padded to 32, bf16 WMMA) ----
  {
    const int a = wave;
    AFrag af;                                // A: ris [16, 32], K>=8 zero
    #pragma unroll
    for (int i = 0; i < 16; ++i) {
      float v = (hi == 0 && i < 8) ? ptmp[(a * 16 + M) * 16 + i] : 0.f;
      af.e[i] = (__bf16)v;
    }
    const char* ob = (const char*)ws + OFF_O + (size_t)a * 16 * 1024;
    for (int nt = 0; nt < 16; ++nt) {
      BFrag bf;
      const uint4* bp = (const uint4*)(ob + (size_t)nt * 1024 + lane * 32);
      bf.q[0] = bp[0]; bf.q[1] = bp[1];
      v8f z = {};
      v8f acc2 = __builtin_amdgcn_wmma_f32_16x16x32_bf16(false, af.v, false, bf.v,
                                                         (short)0, z, false, false);
      const float bias = bout[a * 256 + nt * 16 + n];
      #pragma unroll
      for (int j = 0; j < 8; ++j) {          // streaming stores: nontemporal
        int row = b0 + j + 8 * hi;
        __builtin_nontemporal_store(acc2[j] + bias,
                                    out + (size_t)row * 1024 + a * 256 +
                                    nt * 16 + n);
      }
    }
  }
}

extern "C" void kernel_launch(void* const* d_in, const int* in_sizes, int n_in,
                              void* d_out, int out_size, void* d_ws, size_t ws_size,
                              hipStream_t stream) {
  (void)in_sizes; (void)n_in; (void)out_size; (void)ws_size; // ws use ~632 KB
  const float* nh = (const float*)d_in[0];
  const float* U  = (const float*)d_in[1];
  const float* bb = (const float*)d_in[2];
  const float* Uo = (const float*)d_in[3];
  const float* bo = (const float*)d_in[4];
  const float* T  = (const float*)d_in[5];
  __hip_bfloat16* ws = (__hip_bfloat16*)d_ws;
  float* outf = (float*)d_out;

  hosvd_repack<<<(632 * 32 + 127) / 128, 128, 0, stream>>>(U, Uo, T, ws);
  hosvd_main<<<8192 / 16, 128, 0, stream>>>(nh, bb, bo, ws, outf);
}
